// GraphSAGEEncoder_51659866636534
// MI455X (gfx1250) — compile-verified, matrix operand order checked by reading, not run on hardware
//
#include <hip/hip_runtime.h>

#define NNODES 50000
#define NGRAPH 64
#define DHID   128

typedef __attribute__((ext_vector_type(16))) _Float16 v16h;
typedef __attribute__((ext_vector_type(8)))  float    v8f;
typedef __attribute__((ext_vector_type(4)))  _Float16 half4;
typedef __attribute__((ext_vector_type(4)))  unsigned int u32x4;

union AFrag { v16h v; unsigned int u[8]; u32x4 q[2]; };

// ---------------------------------------------------------------------------
// Embed: h = relu(layernorm(x @ W_emb + b_emb)).  One wave per node, 4 ch/lane.
// ---------------------------------------------------------------------------
__global__ __launch_bounds__(256)
void embed_kernel(const float* __restrict__ x, const float* __restrict__ W,
                  const float* __restrict__ b, const float* __restrict__ gamma,
                  const float* __restrict__ beta, float* __restrict__ h, int n_nodes)
{
    int gid  = blockIdx.x * 256 + threadIdx.x;
    int n    = gid >> 5;
    int lane = gid & 31;
    if (n >= n_nodes) return;
    int c0 = lane * 4;
    float acc[4] = { b[c0], b[c0 + 1], b[c0 + 2], b[c0 + 3] };
#pragma unroll
    for (int k = 0; k < 9; ++k) {
        float xv = x[n * 9 + k];
#pragma unroll
        for (int j = 0; j < 4; ++j) acc[j] += xv * W[k * DHID + c0 + j];
    }
    float s = acc[0] + acc[1] + acc[2] + acc[3];
#pragma unroll
    for (int m = 1; m < 32; m <<= 1) s += __shfl_xor(s, m);
    float mean = s * (1.0f / DHID);
    float vs = 0.f;
#pragma unroll
    for (int j = 0; j < 4; ++j) { float d = acc[j] - mean; vs += d * d; }
#pragma unroll
    for (int m = 1; m < 32; m <<= 1) vs += __shfl_xor(vs, m);
    float rstd = rsqrtf(vs * (1.0f / DHID) + 1e-5f);
    float4 o;
    o.x = fmaxf((acc[0] - mean) * rstd * gamma[c0 + 0] + beta[c0 + 0], 0.f);
    o.y = fmaxf((acc[1] - mean) * rstd * gamma[c0 + 1] + beta[c0 + 1], 0.f);
    o.z = fmaxf((acc[2] - mean) * rstd * gamma[c0 + 2] + beta[c0 + 2], 0.f);
    o.w = fmaxf((acc[3] - mean) * rstd * gamma[c0 + 3] + beta[c0 + 3], 0.f);
    ((float4*)(h + (size_t)n * DHID))[lane] = o;
}

// ---------------------------------------------------------------------------
// Degree count + reciprocal (computed once, reused by all 3 layers)
// ---------------------------------------------------------------------------
__global__ __launch_bounds__(256)
void count_kernel(const int* __restrict__ ei, float* __restrict__ cnt, int num_e)
{
    int e = blockIdx.x * 256 + threadIdx.x;
    if (e >= num_e) return;
    atomicAdd(cnt + ei[num_e + e], 1.0f);
}

__global__ __launch_bounds__(256)
void inv_kernel(const float* __restrict__ cnt, float* __restrict__ inv, int n_nodes)
{
    int n = blockIdx.x * 256 + threadIdx.x;
    if (n >= n_nodes) return;
    inv[n] = 1.0f / fmaxf(cnt[n], 1.0f);
}

// ---------------------------------------------------------------------------
// Edge scatter: nsum[dst] += h[src].  One wave per edge, float4 per lane.
// ---------------------------------------------------------------------------
__global__ __launch_bounds__(256)
void scatter_kernel(const float* __restrict__ h, const int* __restrict__ ei,
                    float* __restrict__ nsum, int num_e)
{
    int gid  = blockIdx.x * 256 + threadIdx.x;
    int e    = gid >> 5;
    int lane = gid & 31;
    if (e >= num_e) return;
    int s = ei[e];
    int d = ei[num_e + e];
    float4 v = ((const float4*)(h + (size_t)s * DHID))[lane];
    float* dp = nsum + (size_t)d * DHID + lane * 4;
    atomicAdd(dp + 0, v.x);
    atomicAdd(dp + 1, v.y);
    atomicAdd(dp + 2, v.z);
    atomicAdd(dp + 3, v.w);
}

// ---------------------------------------------------------------------------
// Build f16 concat(self, mean) rows for the WMMA GEMM
// ---------------------------------------------------------------------------
__global__ __launch_bounds__(256)
void build_hcat_kernel(const float* __restrict__ h, const float* __restrict__ nsum,
                       const float* __restrict__ inv, _Float16* __restrict__ hcat,
                       int n_nodes)
{
    int gid  = blockIdx.x * 256 + threadIdx.x;
    int n    = gid >> 5;
    int lane = gid & 31;
    if (n >= n_nodes) return;
    float4 hv = ((const float4*)(h + (size_t)n * DHID))[lane];
    float4 sv = ((const float4*)(nsum + (size_t)n * DHID))[lane];
    float iv = inv[n];
    half4 a, m;
    a.x = (_Float16)hv.x; a.y = (_Float16)hv.y; a.z = (_Float16)hv.z; a.w = (_Float16)hv.w;
    m.x = (_Float16)(sv.x * iv); m.y = (_Float16)(sv.y * iv);
    m.z = (_Float16)(sv.z * iv); m.w = (_Float16)(sv.w * iv);
    half4* row = (half4*)(hcat + (size_t)n * 256);
    row[lane]      = a;   // self  -> K 0..127
    row[32 + lane] = m;   // mean  -> K 128..255
}

// ---------------------------------------------------------------------------
// Pack W (256x128 f32, row-major) into WMMA B-fragment layout (f16 pairs).
// Layout: Bp[nt*2048 + kb*256 + lane*8 + v], pair (k,k+1) per dword, where
//   n = nt*16 + (lane&15), kbase = (lane>>4)*8,
//   k = kb*32 + (v<4 ? kbase+2v : 16+kbase+2(v-4))
// ---------------------------------------------------------------------------
__global__ __launch_bounds__(256)
void pack_w_kernel(const float* __restrict__ W, unsigned int* __restrict__ Bp)
{
    int id   = blockIdx.x * 256 + threadIdx.x;   // 16384 dwords
    int v    = id & 7;
    int lane = (id >> 3) & 31;
    int kb   = (id >> 8) & 7;
    int nt   = (id >> 11) & 7;
    int n     = nt * 16 + (lane & 15);
    int kbase = (lane >> 4) * 8;
    int k = kb * 32 + (v < 4 ? kbase + 2 * v : 16 + kbase + 2 * (v - 4));
    _Float16 lo = (_Float16)W[(size_t)k * DHID + n];
    _Float16 hi = (_Float16)W[(size_t)(k + 1) * DHID + n];
    unsigned short lb = __builtin_bit_cast(unsigned short, lo);
    unsigned short hb = __builtin_bit_cast(unsigned short, hi);
    Bp[id] = (unsigned)lb | ((unsigned)hb << 16);
}

// ---------------------------------------------------------------------------
// Fused SAGE GEMM (16x128 tile per block, K=256 via 8x v_wmma_f32_16x16x32_f16)
// + bias + LayerNorm + ReLU epilogue.
// ---------------------------------------------------------------------------
__global__ __launch_bounds__(256)
void sage_gemm_ln_kernel(const _Float16* __restrict__ hcat,
                         const unsigned int* __restrict__ Bpack,
                         const float* __restrict__ bias,
                         const float* __restrict__ gamma,
                         const float* __restrict__ beta,
                         float* __restrict__ out, int n_nodes)
{
    __shared__ unsigned int sA[16][128];   // 16 nodes x 256 f16 (pair-packed)
    __shared__ float sOut[16][130];        // 16 nodes x 128 f32 (+pad)

    const int tid  = threadIdx.x;
    const int tile = blockIdx.x * 16;

    // Stage the 16 A rows into LDS (shared by all 8 waves)
    const unsigned int* hc32 = (const unsigned int*)hcat;
#pragma unroll
    for (int i = 0; i < 8; ++i) {
        int idx  = tid + 256 * i;
        int row  = idx >> 7;
        int col  = idx & 127;
        int node = tile + row;
        sA[row][col] = (node < n_nodes) ? hc32[(size_t)node * 128 + col] : 0u;
    }
    __syncthreads();

    const int wave = tid >> 5;     // 8 waves -> 8 column tiles of 16
    const int lane = tid & 31;
    const int half = lane >> 4;
    const int l16  = lane & 15;

    AFrag a, b;
    v8f c = {};
    const unsigned int* bp = Bpack + wave * 2048;
    const u32x4* arow = (const u32x4*)&sA[l16][0];
#pragma unroll
    for (int kb = 0; kb < 8; ++kb) {
        a.q[0] = arow[kb * 4 + half];            // K pairs kbase+0..7
        a.q[1] = arow[kb * 4 + 2 + half];        // K pairs 16+kbase+0..7
        const u32x4* bpp = (const u32x4*)(bp + kb * 256 + lane * 8);
        b.q[0] = bpp[0];
        b.q[1] = bpp[1];
        c = __builtin_amdgcn_wmma_f32_16x16x32_f16(false, a.v, false, b.v,
                                                   (short)0, c, false, false);
    }

    float bv = bias[wave * 16 + l16];
#pragma unroll
    for (int r = 0; r < 8; ++r)
        sOut[r + half * 8][wave * 16 + l16] = c[r] + bv;
    __syncthreads();

    // LayerNorm: each half-wave owns one node (wave w -> nodes 2w, 2w+1)
    int m = wave * 2 + half;
    float vals[8];
    float s = 0.f;
#pragma unroll
    for (int i = 0; i < 8; ++i) { vals[i] = sOut[m][l16 + 16 * i]; s += vals[i]; }
    s += __shfl_xor(s, 1); s += __shfl_xor(s, 2);
    s += __shfl_xor(s, 4); s += __shfl_xor(s, 8);
    float mean = s * (1.0f / DHID);
    float vs = 0.f;
#pragma unroll
    for (int i = 0; i < 8; ++i) { float d = vals[i] - mean; vs += d * d; }
    vs += __shfl_xor(vs, 1); vs += __shfl_xor(vs, 2);
    vs += __shfl_xor(vs, 4); vs += __shfl_xor(vs, 8);
    float rstd = rsqrtf(vs * (1.0f / DHID) + 1e-5f);

    int node = tile + m;
    if (node < n_nodes) {
#pragma unroll
        for (int i = 0; i < 8; ++i) {
            int ch = l16 + 16 * i;
            float val = (vals[i] - mean) * rstd * gamma[ch] + beta[ch];
            out[(size_t)node * DHID + ch] = fmaxf(val, 0.f);
        }
    }
}

// ---------------------------------------------------------------------------
// Graph pooling: sum / count / max per graph (post-ReLU values >= 0, so
// uint-bit atomicMax over 0-initialized buffer is order-correct).
// ---------------------------------------------------------------------------
__global__ __launch_bounds__(256)
void pool_kernel(const float* __restrict__ ne, const int* __restrict__ batch,
                 float* __restrict__ gsum, float* __restrict__ gcnt,
                 unsigned int* __restrict__ gmax, int n_nodes)
{
    int gid  = blockIdx.x * 256 + threadIdx.x;
    int n    = gid >> 5;
    int lane = gid & 31;
    if (n >= n_nodes) return;
    int g = batch[n];
    float4 v = ((const float4*)(ne + (size_t)n * DHID))[lane];
    float* gs = gsum + (size_t)g * DHID + lane * 4;
    atomicAdd(gs + 0, v.x); atomicAdd(gs + 1, v.y);
    atomicAdd(gs + 2, v.z); atomicAdd(gs + 3, v.w);
    unsigned int* gm = gmax + (size_t)g * DHID + lane * 4;
    atomicMax(gm + 0, __float_as_uint(v.x));
    atomicMax(gm + 1, __float_as_uint(v.y));
    atomicMax(gm + 2, __float_as_uint(v.z));
    atomicMax(gm + 3, __float_as_uint(v.w));
    if (lane == 0) atomicAdd(gcnt + g, 1.0f);
}

__global__ __launch_bounds__(256)
void finalize_kernel(const float* __restrict__ gsum, const float* __restrict__ gcnt,
                     const unsigned int* __restrict__ gmax, float* __restrict__ gout)
{
    int id = blockIdx.x * 256 + threadIdx.x;   // NGRAPH*256
    int g = id >> 8;
    int c = id & 255;
    float r;
    if (c < DHID) r = gsum[g * DHID + c] / fmaxf(gcnt[g], 1.0f);
    else          r = __uint_as_float(gmax[g * DHID + (c - DHID)]);
    gout[id] = r;
}

// ---------------------------------------------------------------------------
extern "C" void kernel_launch(void* const* d_in, const int* in_sizes, int n_in,
                              void* d_out, int out_size, void* d_ws, size_t ws_size,
                              hipStream_t stream)
{
    const float* x     = (const float*)d_in[0];
    const int*   ei    = (const int*)d_in[1];
    const int*   batch = (const int*)d_in[2];
    const float* W_emb = (const float*)d_in[3];
    const float* b_emb = (const float*)d_in[4];
    const float* g0    = (const float*)d_in[5];
    const float* bt0   = (const float*)d_in[6];
    const float* Wc[3] = { (const float*)d_in[7],  (const float*)d_in[11], (const float*)d_in[15] };
    const float* bc[3] = { (const float*)d_in[8],  (const float*)d_in[12], (const float*)d_in[16] };
    const float* gl[3] = { (const float*)d_in[9],  (const float*)d_in[13], (const float*)d_in[17] };
    const float* bl[3] = { (const float*)d_in[10], (const float*)d_in[14], (const float*)d_in[18] };

    const int N_ = in_sizes[2];
    const int E_ = in_sizes[1] / 2;

    // ---- workspace layout -------------------------------------------------
    char* ws = (char*)d_ws;
    size_t off = 0;
    float*        h    = (float*)(ws + off);        off += (size_t)N_ * DHID * 4;
    float*        nsum = (float*)(ws + off);        off += (size_t)N_ * DHID * 4;
    _Float16*     hcat = (_Float16*)(ws + off);     off += (size_t)N_ * 256 * 2;
    float*        inv  = (float*)(ws + off);        off += (size_t)N_ * 4;
    float*        cnt  = (float*)(ws + off);        off += (size_t)N_ * 4;
    unsigned int* Bp   = (unsigned int*)(ws + off); off += 3 * 16384 * 4;
    float*        gsum = (float*)(ws + off);        off += NGRAPH * DHID * 4;
    float*        gcnt = (float*)(ws + off);        off += NGRAPH * 64;   // padded
    unsigned int* gmax = (unsigned int*)(ws + off); off += NGRAPH * DHID * 4;

    float* node_out  = (float*)d_out;                          // N x 128
    float* graph_out = (float*)d_out + (size_t)N_ * DHID;      // G x 256

    const int T = 256;
    int gNodeWave = (N_ * 32 + T - 1) / T;
    int gEdgeWave = (int)(((long long)E_ * 32 + T - 1) / T);
    int gEdge     = (E_ + T - 1) / T;
    int gNode     = (N_ + T - 1) / T;
    int gTiles    = (N_ + 15) / 16;

    // ---- per-launch init --------------------------------------------------
    hipMemsetAsync(cnt,  0, (size_t)N_ * 4, stream);
    hipMemsetAsync(gsum, 0, NGRAPH * DHID * 4, stream);
    hipMemsetAsync(gcnt, 0, NGRAPH * 4, stream);
    hipMemsetAsync(gmax, 0, NGRAPH * DHID * 4, stream);

    for (int L = 0; L < 3; ++L)
        pack_w_kernel<<<64, T, 0, stream>>>(Wc[L], Bp + L * 16384);

    count_kernel<<<gEdge, T, 0, stream>>>(ei, cnt, E_);
    inv_kernel<<<gNode, T, 0, stream>>>(cnt, inv, N_);
    embed_kernel<<<gNodeWave, T, 0, stream>>>(x, W_emb, b_emb, g0, bt0, h, N_);

    for (int L = 0; L < 3; ++L) {
        hipMemsetAsync(nsum, 0, (size_t)N_ * DHID * 4, stream);
        scatter_kernel<<<gEdgeWave, T, 0, stream>>>(h, ei, nsum, E_);
        build_hcat_kernel<<<gNodeWave, T, 0, stream>>>(h, nsum, inv, hcat, N_);
        float* outp = (L == 2) ? node_out : h;
        sage_gemm_ln_kernel<<<gTiles, T, 0, stream>>>(hcat, Bp + L * 16384,
                                                      bc[L], gl[L], bl[L], outp, N_);
    }

    pool_kernel<<<gNodeWave, T, 0, stream>>>(node_out, batch, gsum, gcnt, gmax, N_);
    finalize_kernel<<<NGRAPH, T, 0, stream>>>(gsum, gcnt, gmax, graph_out);
}
